// StatefulStereoMaskNet_35081292874134
// MI455X (gfx1250) — compile-verified, matrix operand order checked by reading, not run on hardware
//
#include <hip/hip_runtime.h>
#include <hip/hip_bf16.h>

typedef __attribute__((ext_vector_type(16))) __bf16 v16bf;
typedef __attribute__((ext_vector_type(8)))  __bf16 v8bf;
typedef __attribute__((ext_vector_type(8)))  float  v8f;

#define B_    32
#define T_    1000
#define FREQ_ 257
#define HID_  320
#define G4_   1280   // 4*HID
#define INP_  800    // 771 padded to mult of 32
#define INS_  771
#define FCN_  320    // 257 padded to mult of 64 (edge-guard-free N blocking)

__device__ __forceinline__ float sigmoidf_(float x) { return 1.f / (1.f + __expf(-x)); }

// Build a 16x32 bf16 WMMA operand fragment for this lane from two contiguous
// 16-byte chunks (ISA 7.12.2 16-bit A/B layout); compiles to 2x b128 loads.
__device__ __forceinline__ v16bf load_frag(const __bf16* __restrict__ p)
{
    v8bf lo = *(const v8bf*)(p);
    v8bf hi = *(const v8bf*)(p + 16);
    v16bf r;
#pragma unroll
    for (int e = 0; e < 8; ++e) { r[e] = lo[e]; r[8 + e] = hi[e]; }
    return r;
}

#define WMMA_BF16(Afrag, Bfrag, Cacc) \
    __builtin_amdgcn_wmma_f32_16x16x32_bf16(false, (Afrag), false, (Bfrag), \
                                            (short)0, (Cacc), false, false)

// ---------------------------------------------------------------------------
// Featurize: [B,F,T] stereo spectra -> bf16 feature rows [B*T, INP_]
// cols: [0,257)=log1p(|L|), [257,514)=cos(ipd), [514,771)=sin(ipd), [771,800)=0
// ---------------------------------------------------------------------------
__global__ void k_featurize(const float* __restrict__ xlr, const float* __restrict__ xli,
                            const float* __restrict__ xrr, const float* __restrict__ xri,
                            __bf16* __restrict__ feats)
{
    long idx = (long)blockIdx.x * blockDim.x + threadIdx.x;
    long n = (long)B_ * FREQ_ * T_;
    if (idx >= n) return;
    int t = (int)(idx % T_);
    int f = (int)((idx / T_) % FREQ_);
    int b = (int)(idx / ((long)T_ * FREQ_));
    long src = ((long)b * FREQ_ + f) * T_ + t;
    float lr = xlr[src], li = xli[src], rr = xrr[src], ri = xri[src];
    float mag = sqrtf(lr * lr + li * li);
    float lg  = log1pf(mag);
    float ipd = atan2f(li, lr) - atan2f(ri, rr);
    long row = (long)b * T_ + t;
    __bf16* fr = feats + row * INP_;
    fr[f]             = (__bf16)lg;
    fr[FREQ_ + f]     = (__bf16)__cosf(ipd);
    fr[2 * FREQ_ + f] = (__bf16)__sinf(ipd);
    if (f < INP_ - 3 * FREQ_)               // zero the 29 pad columns
        fr[3 * FREQ_ + f] = (__bf16)0.f;
}

// ---------------------------------------------------------------------------
// f32 -> bf16 with zero padding (weights into [rows_dst, cols_dst])
// ---------------------------------------------------------------------------
__global__ void k_convert_pad(__bf16* __restrict__ dst, const float* __restrict__ src,
                              int rows_dst, int cols_dst, int rows_src, int cols_src)
{
    long idx = (long)blockIdx.x * blockDim.x + threadIdx.x;
    long total = (long)rows_dst * cols_dst;
    if (idx >= total) return;
    int c = (int)(idx % cols_dst);
    int r = (int)(idx / cols_dst);
    float v = (r < rows_src && c < cols_src) ? src[(long)r * cols_src + c] : 0.f;
    dst[idx] = (__bf16)v;
}

// ---------------------------------------------------------------------------
// WMMA GEMM, 1x4 tile register blocking per wave, K compile-time + full unroll
// (SSA renaming -> load/WMMA software pipelining with no buffer-copy movs):
//   C[M,N] = A[M,K](bf16 row-major) x B[N,K](bf16 row-major; = K x N col-major
//   operand) + bias0[col] + bias1[col]
// 128 threads = 4 waves; each wave computes a 16x64 output slab, one A
// fragment feeding 4 WMMAs per k-step. N must be a multiple of 64, K of 32.
// ---------------------------------------------------------------------------
template <int K>
__global__ void __launch_bounds__(128)
k_gemm_bf16(const __bf16* __restrict__ A, const __bf16* __restrict__ Bm,
            float* __restrict__ C, int M, int N,
            const float* __restrict__ bias0,
            const float* __restrict__ bias1, int biasN)
{
    int wave = threadIdx.x >> 5;
    int lane = threadIdx.x & 31;
    int ngrp = N >> 6;                                 // groups of 4 N-tiles
    long total = (long)(M >> 4) * ngrp;
    long job = (long)blockIdx.x * 4 + wave;
    if (job >= total) return;                          // wave-uniform exit
    int  gn = (int)(job % ngrp);
    long tm = job / ngrp;
    int lrow  = lane & 15;
    int khalf = (lane & 16) ? 8 : 0;
    int mhalf = (lane & 16) ? 8 : 0;
    const __bf16* Ap = A  + ((long)tm * 16 + lrow) * K + khalf;
    const __bf16* Bp = Bm + ((long)gn * 64 + lrow) * K + khalf;
    v8f acc[4];
#pragma unroll
    for (int j = 0; j < 4; ++j) acc[j] = (v8f){};
#pragma unroll
    for (int k0 = 0; k0 < K; k0 += 32) {
        v16bf af = load_frag(Ap + k0);                 // loaded once, used 4x
#pragma unroll
        for (int j = 0; j < 4; ++j)
            acc[j] = WMMA_BF16(af, load_frag(Bp + (long)j * 16 * K + k0), acc[j]);
    }
#pragma unroll
    for (int j = 0; j < 4; ++j) {
        int col = gn * 64 + j * 16 + lrow;
        float bias = 0.f;
        if (col < biasN) {
            if (bias0) bias += bias0[col];
            if (bias1) bias += bias1[col];
        }
#pragma unroll
        for (int r = 0; r < 8; ++r) {
            long row = tm * 16 + r + mhalf;
            C[row * (long)N + col] = acc[j][r] + bias;
        }
    }
}

// ---------------------------------------------------------------------------
// Persistent LSTM recurrence, one workgroup per 16-batch M-tile (grid = 2).
// The recurrence is independent per batch row, so the two blocks never
// communicate and run on separate WGPs, halving per-step latency.
// 512 threads = 16 waves (4/SIMD); each wave owns 5 N-tiles (tn = wave+16i),
// 50 WMMAs/step, acc[5]=40 VGPRs -> no spill pressure. A memory clobber at
// the top of each t-iteration stops the compiler from hoisting the
// t-invariant W_hh fragment loads (which previously spilled to scratch);
// they stay global_load_b128 served from L2 (W_hh bf16 = 0.8 MB, resident).
// State h/c + staged gates live in LDS (110 KB < 320 KB/WGP).
// xg  : [B*T, 1280] f32 (precomputed input projection + both biases)
// Whh : [1280, 320] bf16 ; hout: [B*T, 320] bf16
// ---------------------------------------------------------------------------
__global__ void __launch_bounds__(512, 1)
k_lstm_scan(const float* __restrict__ xg, const __bf16* __restrict__ Whh,
            __bf16* __restrict__ hout)
{
    __shared__ __align__(16) float  s_gates[16 * G4_];   // 80 KB
    __shared__ __align__(16) float  s_c[16 * HID_];      // 20 KB
    __shared__ __align__(16) __bf16 s_h[16 * HID_];      // 10 KB

    int tid  = threadIdx.x;
    int wave = tid >> 5, lane = tid & 31;
    int lrow  = lane & 15;
    int khalf = (lane & 16) ? 8 : 0;
    int mhalf = (lane & 16) ? 8 : 0;
    int bbase = blockIdx.x * 16;                 // batch rows of this block

    for (int u = tid; u < 16 * HID_; u += 512) { s_c[u] = 0.f; s_h[u] = (__bf16)0.f; }
    __syncthreads();

    const __bf16* Arow = s_h + lrow * HID_ + khalf;
    const __bf16* Brow = Whh + ((long)wave * 16 + lrow) * HID_ + khalf;

    for (int t = 0; t < T_; ++t) {
        asm volatile("" ::: "memory");           // keep W_hh loads in-loop
        v8f acc[5];
#pragma unroll
        for (int i = 0; i < 5; ++i) acc[i] = (v8f){};
        // ---- gates = h_prev @ Whh^T : k-outer, 5 N-tiles inner, B prefetch
#pragma unroll
        for (int kk = 0; kk < 10; ++kk) {
            v16bf af  = load_frag(Arow + kk * 32);            // 1 LDS load / k
            v16bf bfc = load_frag(Brow + kk * 32);            // tile i = 0
#pragma unroll
            for (int i = 0; i < 5; ++i) {
                v16bf bfn;
                if (i < 4)                                    // prefetch tile i+1
                    bfn = load_frag(Brow + (long)(i + 1) * 16 * 16 * HID_ + kk * 32);
                acc[i] = WMMA_BF16(af, bfc, acc[i]);
                if (i < 4) bfc = bfn;
            }
        }
        // ---- add precomputed input projection, stage gates in LDS
#pragma unroll
        for (int i = 0; i < 5; ++i) {
            int col = (wave + 16 * i) * 16 + lrow;
#pragma unroll
            for (int r = 0; r < 8; ++r) {
                int brow = r + mhalf;                         // 0..15 local
                s_gates[brow * G4_ + col] =
                    acc[i][r] + xg[((long)(bbase + brow) * T_ + t) * G4_ + col];
            }
        }
        __syncthreads();
        // ---- LSTM cell update (PyTorch gate order i,f,g,o), f32 state
        for (int u = tid; u < 16 * HID_; u += 512) {
            int b = u / HID_, j = u % HID_;
            const float* g = s_gates + b * G4_;
            float ig = sigmoidf_(g[j]);
            float fg = sigmoidf_(g[HID_ + j]);
            float gg = tanhf(g[2 * HID_ + j]);
            float og = sigmoidf_(g[3 * HID_ + j]);
            float c = fg * s_c[u] + ig * gg;
            float h = og * tanhf(c);
            s_c[u] = c;
            __bf16 hb = (__bf16)h;
            s_h[u] = hb;
            hout[((long)(bbase + b) * T_ + t) * HID_ + j] = hb;
        }
        __syncthreads();
    }
}

// ---------------------------------------------------------------------------
// Epilogue: mask = sigmoid(logits), outputs sep_real|sep_imag|mask in [B,F,T]
// ---------------------------------------------------------------------------
__global__ void k_epilogue(const float* __restrict__ logits,   // [B*T, FCN_]
                           const float* __restrict__ xlr, const float* __restrict__ xli,
                           float* __restrict__ out)
{
    long idx = (long)blockIdx.x * blockDim.x + threadIdx.x;
    long S = (long)B_ * FREQ_ * T_;
    if (idx >= S) return;
    int t = (int)(idx % T_);
    int f = (int)((idx / T_) % FREQ_);
    int b = (int)(idx / ((long)T_ * FREQ_));
    float m = sigmoidf_(logits[((long)b * T_ + t) * FCN_ + f]);
    long s = ((long)b * FREQ_ + f) * T_ + t;
    out[s]         = xlr[s] * m;
    out[S + s]     = xli[s] * m;
    out[2 * S + s] = m;
}

// ---------------------------------------------------------------------------
extern "C" void kernel_launch(void* const* d_in, const int* in_sizes, int n_in,
                              void* d_out, int out_size, void* d_ws, size_t ws_size,
                              hipStream_t stream)
{
    const float* xlr  = (const float*)d_in[0];
    const float* xli  = (const float*)d_in[1];
    const float* xrr  = (const float*)d_in[2];
    const float* xri  = (const float*)d_in[3];
    const float* Wih0 = (const float*)d_in[4];
    const float* Whh0 = (const float*)d_in[5];
    const float* bih0 = (const float*)d_in[6];
    const float* bhh0 = (const float*)d_in[7];
    const float* Wih1 = (const float*)d_in[8];
    const float* Whh1 = (const float*)d_in[9];
    const float* bih1 = (const float*)d_in[10];
    const float* bhh1 = (const float*)d_in[11];
    const float* fcW  = (const float*)d_in[12];
    const float* fcb  = (const float*)d_in[13];

    // workspace carve-out (256B aligned chunks)
    size_t off = 0;
    auto carve = [&](size_t bytes) -> void* {
        void* p = (char*)d_ws + off;
        off += (bytes + 255) & ~(size_t)255;
        return p;
    };
    __bf16* feats  = (__bf16*)carve((size_t)B_ * T_ * INP_ * 2);   // 51.2 MB
    __bf16* wih0p  = (__bf16*)carve((size_t)G4_ * INP_ * 2);       //  2.0 MB
    __bf16* whh0b  = (__bf16*)carve((size_t)G4_ * HID_ * 2);       //  0.8 MB
    __bf16* wih1b  = (__bf16*)carve((size_t)G4_ * HID_ * 2);       //  0.8 MB
    __bf16* whh1b  = (__bf16*)carve((size_t)G4_ * HID_ * 2);       //  0.8 MB
    __bf16* fcwp   = (__bf16*)carve((size_t)FCN_ * HID_ * 2);      //  0.2 MB
    __bf16* h1     = (__bf16*)carve((size_t)B_ * T_ * HID_ * 2);   // 20.5 MB
    __bf16* h2     = (__bf16*)carve((size_t)B_ * T_ * HID_ * 2);   // 20.5 MB
    float*  xgbuf  = (float*)carve((size_t)B_ * T_ * G4_ * 4);     // 163.8 MB
    float*  logits = (float*)feats;  // reuse: feats dead after GEMM1 (41<=51.2 MB)

    const long NPT = (long)B_ * FREQ_ * T_;
    const int  TB  = 256;
    dim3 blk(TB);

    // 1) features
    k_featurize<<<dim3((unsigned)((NPT + TB - 1) / TB)), blk, 0, stream>>>(
        xlr, xli, xrr, xri, feats);

    // 2) weight conversions (f32 -> bf16, padded)
    k_convert_pad<<<dim3((G4_ * INP_ + TB - 1) / TB), blk, 0, stream>>>(
        wih0p, Wih0, G4_, INP_, G4_, INS_);
    k_convert_pad<<<dim3((G4_ * HID_ + TB - 1) / TB), blk, 0, stream>>>(
        whh0b, Whh0, G4_, HID_, G4_, HID_);
    k_convert_pad<<<dim3((G4_ * HID_ + TB - 1) / TB), blk, 0, stream>>>(
        wih1b, Wih1, G4_, HID_, G4_, HID_);
    k_convert_pad<<<dim3((G4_ * HID_ + TB - 1) / TB), blk, 0, stream>>>(
        whh1b, Whh1, G4_, HID_, G4_, HID_);
    k_convert_pad<<<dim3((FCN_ * HID_ + TB - 1) / TB), blk, 0, stream>>>(
        fcwp, fcW, FCN_, HID_, FREQ_, HID_);

    const int M = B_ * T_;                    // 32000 rows
    // 3) layer-0 input projection: xg0 = feats @ Wih0^T + bih0 + bhh0
    {
        long jobs = (long)(M / 16) * (G4_ / 64);           // 40000
        k_gemm_bf16<INP_><<<dim3((unsigned)((jobs + 3) / 4)), dim3(128), 0, stream>>>(
            feats, wih0p, xgbuf, M, G4_, bih0, bhh0, G4_);
    }
    // 4) layer-0 recurrence (2 independent batch halves)
    k_lstm_scan<<<dim3(2), dim3(512), 0, stream>>>(xgbuf, whh0b, h1);

    // 5) layer-1 input projection: xg1 = h1 @ Wih1^T + bih1 + bhh1 (reuse xgbuf)
    {
        long jobs = (long)(M / 16) * (G4_ / 64);
        k_gemm_bf16<HID_><<<dim3((unsigned)((jobs + 3) / 4)), dim3(128), 0, stream>>>(
            h1, wih1b, xgbuf, M, G4_, bih1, bhh1, G4_);
    }
    // 6) layer-1 recurrence
    k_lstm_scan<<<dim3(2), dim3(512), 0, stream>>>(xgbuf, whh1b, h2);

    // 7) FC: logits = h2 @ fcW^T + fcb  (N padded to 320)
    {
        long jobs = (long)(M / 16) * (FCN_ / 64);          // 10000
        k_gemm_bf16<HID_><<<dim3((unsigned)((jobs + 3) / 4)), dim3(128), 0, stream>>>(
            h2, fcwp, logits, M, FCN_, fcb, nullptr, FREQ_);
    }
    // 8) mask + separation outputs
    k_epilogue<<<dim3((unsigned)((NPT + TB - 1) / TB)), blk, 0, stream>>>(
        logits, xlr, xli, (float*)d_out);
}
